// DMM_41755672052461
// MI455X (gfx1250) — matrix operand order repaired and checked by reading, not compiled, not run on hardware
//
#include <hip/hip_runtime.h>
#include <cstdint>

#define T_DIM 512
#define B_DIM 256
#define OBS_DIM 128
#define LAT_DIM 64
#define HID_DIM 256
#define G_DIM (3 * HID_DIM)

typedef __attribute__((ext_vector_type(16))) __bf16 v16bf;
typedef __attribute__((ext_vector_type(8)))  float  f32x8;

union BfVec { v16bf v; uint32_t u[8]; };

__device__ __forceinline__ f32x8 zero8() {
  f32x8 z = {0.f, 0.f, 0.f, 0.f, 0.f, 0.f, 0.f, 0.f};
  return z;
}

// round-to-nearest-even f32 -> bf16, packed pair
__device__ __forceinline__ uint32_t pack_bf2(float a, float b) {
  uint32_t ua = __float_as_uint(a), ub = __float_as_uint(b);
  ua += 0x7FFFu + ((ua >> 16) & 1u);
  ub += 0x7FFFu + ((ub >> 16) & 1u);
  return (ua >> 16) | (ub & 0xFFFF0000u);
}

// A operand: 16x32 bf16 tile from row-major f32 (global or LDS), row stride ldr.
// Lane l row=l&15; VGPR v holds K = (l>=16?16:0)+2*v (+1 in hi half).
__device__ __forceinline__ v16bf loadA_f32(const float* base, int ldr, int k0) {
  int l = threadIdx.x & 31;
  const float* p = base + (size_t)(l & 15) * ldr + k0 + ((l >> 4) << 4);
  BfVec r;
#pragma unroll
  for (int v = 0; v < 8; ++v) {
    int off = (v << 1);
    r.u[v] = pack_bf2(p[off], p[off + 1]);
  }
  return r.v;
}

// B operand: 32x16 bf16 tile. Weight stored row-major (out N, in K) bf16 == [N][K],
// lane l col n0+(l&15); VGPR v holds K = k0+(l>=16?16:0)+2v,+1 -> one dword load each.
__device__ __forceinline__ v16bf loadB_w(const uint16_t* W, int ldk, int n0, int k0) {
  int l = threadIdx.x & 31;
  const uint32_t* p =
      (const uint32_t*)(W + (size_t)(n0 + (l & 15)) * ldk + k0 + ((l >> 4) << 4));
  BfVec r;
#pragma unroll
  for (int v = 0; v < 8; ++v) r.u[v] = p[v];
  return r.v;
}

__device__ __forceinline__ f32x8 wmma_bf(v16bf a, v16bf b, f32x8 c) {
  return __builtin_amdgcn_wmma_f32_16x16x32_bf16(false, a, false, b, (short)0, c,
                                                 false, false);
}

// D (16x16 f32) scatter: lane l col n0+(l&15); VGPR v row = v + (l>=16?8:0)
__device__ __forceinline__ void storeD(float* base, int ldr, int n0, f32x8 acc) {
  int l = threadIdx.x & 31;
  int n = n0 + (l & 15), mb = (l >> 4) << 3;
#pragma unroll
  for (int v = 0; v < 8; ++v) base[(size_t)(v + mb) * ldr + n] = acc[v];
}

__device__ __forceinline__ float sigmoidf_(float x) { return 1.f / (1.f + __expf(-x)); }

// ---------------------------------------------------------------------------
// init helpers
// ---------------------------------------------------------------------------
__global__ void cvt_bf16_kernel(const float* __restrict__ s, uint16_t* __restrict__ d,
                                int n) {
  int i = blockIdx.x * blockDim.x + threadIdx.x;
  if (i < n) {
    uint32_t u = __float_as_uint(s[i]);
    u += 0x7FFFu + ((u >> 16) & 1u);
    d[i] = (uint16_t)(u >> 16);
  }
}

__global__ void zero_f32_kernel(float* __restrict__ p, int n) {
  int i = blockIdx.x * blockDim.x + threadIdx.x;
  if (i < n) p[i] = 0.f;
}

// ---------------------------------------------------------------------------
// GRU step: one WG per 16-row batch band. C[16x768] = x_t@Wih^T + h@Whh^T via
// WMMA into LDS (n-gate x/h parts kept separate for r*hn), then fused gating.
// ---------------------------------------------------------------------------
__global__ __launch_bounds__(256) void gru_step_kernel(
    const float* __restrict__ x_t, const uint16_t* __restrict__ Wih,
    const uint16_t* __restrict__ Whh, const float* __restrict__ b_ih,
    const float* __restrict__ b_hh, float* __restrict__ h) {
  __shared__ float Crz[16 * 512];
  __shared__ float Cxn[16 * 256];
  __shared__ float Chn[16 * 256];
  const int b0 = blockIdx.x * 16;
  // wave index as an SGPR: task-loop branches become scalar, EXEC stays all-1s
  // at every WMMA (ISA requirement) and no v_cmpx/exec-mask churn is emitted.
  const int wave = __builtin_amdgcn_readfirstlane(threadIdx.x >> 5);

  v16bf Ax[4], Ah[8];
#pragma unroll
  for (int c = 0; c < 4; ++c)
    Ax[c] = loadA_f32(x_t + (size_t)b0 * OBS_DIM, OBS_DIM, c * 32);
#pragma unroll
  for (int c = 0; c < 8; ++c)
    Ah[c] = loadA_f32(h + (size_t)b0 * HID_DIM, HID_DIM, c * 32);

  for (int task = wave; task < 48; task += 8) {
    if (task < 32) {  // r,z gates: x-part + h-part combined
      int n0 = task * 16;
      f32x8 acc = zero8();
#pragma unroll
      for (int c = 0; c < 4; ++c) acc = wmma_bf(Ax[c], loadB_w(Wih, OBS_DIM, n0, c * 32), acc);
#pragma unroll
      for (int c = 0; c < 8; ++c) acc = wmma_bf(Ah[c], loadB_w(Whh, HID_DIM, n0, c * 32), acc);
      storeD(Crz, 512, n0, acc);
    } else {  // n gate: keep x and h contributions separate
      int nl = (task - 32) * 16;
      int n0 = 512 + nl;
      f32x8 ax = zero8(), ah = zero8();
#pragma unroll
      for (int c = 0; c < 4; ++c) ax = wmma_bf(Ax[c], loadB_w(Wih, OBS_DIM, n0, c * 32), ax);
#pragma unroll
      for (int c = 0; c < 8; ++c) ah = wmma_bf(Ah[c], loadB_w(Whh, HID_DIM, n0, c * 32), ah);
      storeD(Cxn, 256, nl, ax);
      storeD(Chn, 256, nl, ah);
    }
  }
  __syncthreads();

  for (int e = threadIdx.x; e < 16 * 256; e += 256) {
    int m = e >> 8, j = e & 255;
    float r  = sigmoidf_(Crz[m * 512 + j]       + b_ih[j]       + b_hh[j]);
    float zg = sigmoidf_(Crz[m * 512 + 256 + j] + b_ih[256 + j] + b_hh[256 + j]);
    float ng = tanhf(Cxn[m * 256 + j] + b_ih[512 + j] +
                     r * (Chn[m * 256 + j] + b_hh[512 + j]));
    size_t idx = (size_t)(b0 + m) * HID_DIM + j;
    h[idx] = (1.f - zg) * ng + zg * h[idx];
  }
}

// ---------------------------------------------------------------------------
// VAE step: prior + posterior MLPs, reparameterize, decoder — all in one WG
// per 16-row band, staged through LDS.
// ---------------------------------------------------------------------------
__global__ __launch_bounds__(256) void vae_step_kernel(
    const float* __restrict__ h, float* __restrict__ z,
    const float* __restrict__ eps_t,
    const uint16_t* __restrict__ tW1, const float* __restrict__ tb1,
    const uint16_t* __restrict__ tW2, const float* __restrict__ tb2,
    const uint16_t* __restrict__ pW1, const float* __restrict__ pb1,
    const uint16_t* __restrict__ pW2, const float* __restrict__ pb2,
    const uint16_t* __restrict__ dW1, const float* __restrict__ db1,
    const uint16_t* __restrict__ dW2, const float* __restrict__ db2,
    float* __restrict__ o_rec, float* __restrict__ o_pm, float* __restrict__ o_pl,
    float* __restrict__ o_qm, float* __restrict__ o_ql) {
  __shared__ float P[16 * 256];   // prior hidden, reused for decoder hidden
  __shared__ float Q[16 * 256];   // posterior hidden
  __shared__ float PO[16 * 128];  // prior mean|logvar
  __shared__ float QO[16 * 128];  // post mean|logvar
  __shared__ float Zt[16 * 64];
  const int b0 = blockIdx.x * 16;
  const int wave = __builtin_amdgcn_readfirstlane(threadIdx.x >> 5);
  const int lane = threadIdx.x & 31;

  v16bf Az[2], Ah[8];
#pragma unroll
  for (int c = 0; c < 2; ++c)
    Az[c] = loadA_f32(z + (size_t)b0 * LAT_DIM, LAT_DIM, c * 32);
#pragma unroll
  for (int c = 0; c < 8; ++c)
    Ah[c] = loadA_f32(h + (size_t)b0 * HID_DIM, HID_DIM, c * 32);

  // Stage A: hidden layers of prior (K=64) and posterior (K=320, concat h|z)
  for (int task = wave; task < 32; task += 8) {
    if (task < 16) {
      int n0 = task * 16;
      f32x8 acc = zero8();
#pragma unroll
      for (int c = 0; c < 2; ++c) acc = wmma_bf(Az[c], loadB_w(tW1, LAT_DIM, n0, c * 32), acc);
      float bias = tb1[n0 + (lane & 15)];
#pragma unroll
      for (int v = 0; v < 8; ++v) acc[v] = fmaxf(acc[v] + bias, 0.f);
      storeD(P, 256, n0, acc);
    } else {
      int n0 = (task - 16) * 16;
      f32x8 acc = zero8();
#pragma unroll
      for (int c = 0; c < 8; ++c)
        acc = wmma_bf(Ah[c], loadB_w(pW1, HID_DIM + LAT_DIM, n0, c * 32), acc);
#pragma unroll
      for (int c = 0; c < 2; ++c)
        acc = wmma_bf(Az[c], loadB_w(pW1, HID_DIM + LAT_DIM, n0, 256 + c * 32), acc);
      float bias = pb1[n0 + (lane & 15)];
#pragma unroll
      for (int v = 0; v < 8; ++v) acc[v] = fmaxf(acc[v] + bias, 0.f);
      storeD(Q, 256, n0, acc);
    }
  }
  __syncthreads();

  // Stage B: output layers (mean|logvar), K=256 from LDS hiddens
  for (int task = wave; task < 16; task += 8) {
    if (task < 8) {
      int n0 = task * 16;
      f32x8 acc = zero8();
#pragma unroll
      for (int c = 0; c < 8; ++c)
        acc = wmma_bf(loadA_f32(P, 256, c * 32), loadB_w(tW2, HID_DIM, n0, c * 32), acc);
      float bias = tb2[n0 + (lane & 15)];
#pragma unroll
      for (int v = 0; v < 8; ++v) acc[v] += bias;
      storeD(PO, 128, n0, acc);
    } else {
      int n0 = (task - 8) * 16;
      f32x8 acc = zero8();
#pragma unroll
      for (int c = 0; c < 8; ++c)
        acc = wmma_bf(loadA_f32(Q, 256, c * 32), loadB_w(pW2, HID_DIM, n0, c * 32), acc);
      float bias = pb2[n0 + (lane & 15)];
#pragma unroll
      for (int v = 0; v < 8; ++v) acc[v] += bias;
      storeD(QO, 128, n0, acc);
    }
  }
  __syncthreads();

  // Stage C: reparameterization + emit 4 latent outputs + new z state
  for (int e = threadIdx.x; e < 16 * 64; e += 256) {
    int m = e >> 6, j = e & 63;
    float pm = PO[m * 128 + j], pl = PO[m * 128 + 64 + j];
    float qm = QO[m * 128 + j], ql = QO[m * 128 + 64 + j];
    size_t oi = (size_t)(b0 + m) * LAT_DIM + j;
    float zt = qm + eps_t[oi] * __expf(0.5f * ql);
    o_pm[oi] = pm;
    o_pl[oi] = pl;
    o_qm[oi] = qm;
    o_ql[oi] = ql;
    Zt[m * 64 + j] = zt;
    z[oi] = zt;
  }
  __syncthreads();

  // Stage D: decoder hidden (K=64, from Zt)
  for (int task = wave; task < 16; task += 8) {
    int n0 = task * 16;
    f32x8 acc = zero8();
#pragma unroll
    for (int c = 0; c < 2; ++c)
      acc = wmma_bf(loadA_f32(Zt, 64, c * 32), loadB_w(dW1, LAT_DIM, n0, c * 32), acc);
    float bias = db1[n0 + (lane & 15)];
#pragma unroll
    for (int v = 0; v < 8; ++v) acc[v] = fmaxf(acc[v] + bias, 0.f);
    storeD(P, 256, n0, acc);  // reuse P
  }
  __syncthreads();

  // Stage E: reconstruction (K=256) -> straight to global
  for (int task = wave; task < 8; task += 8) {
    int n0 = task * 16;
    f32x8 acc = zero8();
#pragma unroll
    for (int c = 0; c < 8; ++c)
      acc = wmma_bf(loadA_f32(P, 256, c * 32), loadB_w(dW2, HID_DIM, n0, c * 32), acc);
    int n = n0 + (lane & 15), mb = (lane >> 4) << 3;
    float bias = db2[n];
#pragma unroll
    for (int v = 0; v < 8; ++v)
      o_rec[(size_t)(b0 + v + mb) * OBS_DIM + n] = acc[v] + bias;
  }
}

// ---------------------------------------------------------------------------
extern "C" void kernel_launch(void* const* d_in, const int* in_sizes, int n_in,
                              void* d_out, int out_size, void* d_ws, size_t ws_size,
                              hipStream_t stream) {
  (void)in_sizes; (void)n_in; (void)out_size; (void)ws_size;
  const float* x    = (const float*)d_in[0];
  const float* eps  = (const float*)d_in[1];
  const float* W_ih = (const float*)d_in[2];
  const float* W_hh = (const float*)d_in[3];
  const float* b_ih = (const float*)d_in[4];
  const float* b_hh = (const float*)d_in[5];
  const float* tW1  = (const float*)d_in[6];
  const float* tb1  = (const float*)d_in[7];
  const float* tW2  = (const float*)d_in[8];
  const float* tb2  = (const float*)d_in[9];
  const float* pW1  = (const float*)d_in[10];
  const float* pb1  = (const float*)d_in[11];
  const float* pW2  = (const float*)d_in[12];
  const float* pb2  = (const float*)d_in[13];
  const float* dW1  = (const float*)d_in[14];
  const float* db1  = (const float*)d_in[15];
  const float* dW2  = (const float*)d_in[16];
  const float* db2  = (const float*)d_in[17];

  char* ws = (char*)d_ws;
  size_t off = 0;
  auto alloc = [&](size_t bytes) -> char* {
    char* p = ws + off;
    off += (bytes + 255) & ~(size_t)255;
    return p;
  };
  uint16_t* wih_b = (uint16_t*)alloc((size_t)G_DIM * OBS_DIM * 2);
  uint16_t* whh_b = (uint16_t*)alloc((size_t)G_DIM * HID_DIM * 2);
  uint16_t* tw1_b = (uint16_t*)alloc((size_t)HID_DIM * LAT_DIM * 2);
  uint16_t* tw2_b = (uint16_t*)alloc((size_t)2 * LAT_DIM * HID_DIM * 2);
  uint16_t* pw1_b = (uint16_t*)alloc((size_t)HID_DIM * (HID_DIM + LAT_DIM) * 2);
  uint16_t* pw2_b = (uint16_t*)alloc((size_t)2 * LAT_DIM * HID_DIM * 2);
  uint16_t* dw1_b = (uint16_t*)alloc((size_t)HID_DIM * LAT_DIM * 2);
  uint16_t* dw2_b = (uint16_t*)alloc((size_t)OBS_DIM * HID_DIM * 2);
  float* hbuf = (float*)alloc((size_t)B_DIM * HID_DIM * 4);
  float* zbuf = (float*)alloc((size_t)B_DIM * LAT_DIM * 4);

  auto cvt = [&](const float* s, uint16_t* d, int n) {
    cvt_bf16_kernel<<<(n + 255) / 256, 256, 0, stream>>>(s, d, n);
  };
  cvt(W_ih, wih_b, G_DIM * OBS_DIM);
  cvt(W_hh, whh_b, G_DIM * HID_DIM);
  cvt(tW1, tw1_b, HID_DIM * LAT_DIM);
  cvt(tW2, tw2_b, 2 * LAT_DIM * HID_DIM);
  cvt(pW1, pw1_b, HID_DIM * (HID_DIM + LAT_DIM));
  cvt(pW2, pw2_b, 2 * LAT_DIM * HID_DIM);
  cvt(dW1, dw1_b, HID_DIM * LAT_DIM);
  cvt(dW2, dw2_b, OBS_DIM * HID_DIM);
  zero_f32_kernel<<<(B_DIM * HID_DIM + 255) / 256, 256, 0, stream>>>(hbuf, B_DIM * HID_DIM);
  zero_f32_kernel<<<(B_DIM * LAT_DIM + 255) / 256, 256, 0, stream>>>(zbuf, B_DIM * LAT_DIM);

  float* out = (float*)d_out;
  float* o_rec = out;
  float* o_pm = out + (size_t)T_DIM * B_DIM * OBS_DIM;
  float* o_pl = o_pm + (size_t)T_DIM * B_DIM * LAT_DIM;
  float* o_qm = o_pl + (size_t)T_DIM * B_DIM * LAT_DIM;
  float* o_ql = o_qm + (size_t)T_DIM * B_DIM * LAT_DIM;

  for (int t = 0; t < T_DIM; ++t) {
    gru_step_kernel<<<16, 256, 0, stream>>>(x + (size_t)t * B_DIM * OBS_DIM, wih_b,
                                            whh_b, b_ih, b_hh, hbuf);
    vae_step_kernel<<<16, 256, 0, stream>>>(
        hbuf, zbuf, eps + (size_t)t * B_DIM * LAT_DIM, tw1_b, tb1, tw2_b, tb2, pw1_b,
        pb1, pw2_b, pb2, dw1_b, db1, dw2_b, db2,
        o_rec + (size_t)t * B_DIM * OBS_DIM, o_pm + (size_t)t * B_DIM * LAT_DIM,
        o_pl + (size_t)t * B_DIM * LAT_DIM, o_qm + (size_t)t * B_DIM * LAT_DIM,
        o_ql + (size_t)t * B_DIM * LAT_DIM);
  }
}
